// MultiHeadAttention_61830349193272
// MI455X (gfx1250) — matrix operand order repaired and checked by reading, not compiled, not run on hardware
//
#include <hip/hip_runtime.h>

typedef __bf16 bf16_t;
typedef bf16_t v16bf __attribute__((ext_vector_type(16)));
typedef float  v8f   __attribute__((ext_vector_type(8)));

#define NB  2
#define NS  2048
#define ND  1024
#define NH  16
#define NDK 64

// float -> bf16 bits (native v_cvt, round-to-nearest-even)
__device__ __forceinline__ unsigned short f2bf_u(float f) {
    __bf16 h = (__bf16)f;
    return __builtin_bit_cast(unsigned short, h);
}

// async global->LDS copy, 16 bytes per lane (ASYNCcnt-tracked, no VGPR round trip)
__device__ __forceinline__ void async_ld_b128(unsigned lds_off, const void* gptr) {
    asm volatile("global_load_async_to_lds_b128 %0, %1, off"
                 :: "v"(lds_off), "v"(gptr) : "memory");
}
__device__ __forceinline__ unsigned lds_addr_of(const void* p) {
    return (unsigned)(uintptr_t)p;   // low 32 bits of generic ptr = LDS offset
}

// ---- WMMA fragment loaders (wave32; contiguous k order, same in A and B) ----
// A 16x32 (MxK): lane m = lane&15. lanes 0-15 get K 0-15, lanes 16-31 get K 16-31.
__device__ __forceinline__ v16bf load_a_frag(const unsigned short* base, int ld) {
    int lane = threadIdx.x & 31;
    int m  = lane & 15;
    int kb = (lane >> 4) * 16;
    union { v16bf v; unsigned short s[16]; } u;
    const unsigned short* rp = base + m * ld;
#pragma unroll
    for (int j = 0; j < 16; ++j) u.s[j] = rp[kb + j];
    return u.v;
}
// B 32x16 (KxN) from a tile stored [n][k] row-major (B-transposed in LDS):
// lane n = lane&15; lanes 0-15 hold K 0-15, lanes 16-31 hold K 16-31.
__device__ __forceinline__ v16bf load_b_frag(const unsigned short* base, int ld) {
    int lane = threadIdx.x & 31;
    int n  = lane & 15;
    int kb = (lane >> 4) * 16;
    union { v16bf v; unsigned short s[16]; } u;
    const unsigned short* rp = base + n * ld + kb;
#pragma unroll
    for (int j = 0; j < 16; ++j) u.s[j] = rp[j];
    return u.v;
}

__device__ __forceinline__ v8f wmma_bf16(v16bf a, v16bf b, v8f c) {
    return __builtin_amdgcn_wmma_f32_16x16x32_bf16(false, a, false, b, (short)0, c, false, false);
}

// =====================================================================
// Kernel 1: fused QKV projection.  Y = X * W^T + b, bf16 out [b][h][s][dk]
// grid = (M/128, N/64, 3), block = 256.  Register-prefetch pipeline:
// next K-step's fp32 tiles are loaded while WMMA works on the current one.
// =====================================================================
__global__ __launch_bounds__(256)
void qkv_proj_kernel(const float* __restrict__ query, const float* __restrict__ key,
                     const float* __restrict__ value,
                     const float* __restrict__ Wq, const float* __restrict__ bq,
                     const float* __restrict__ Wk, const float* __restrict__ bk,
                     const float* __restrict__ Wv, const float* __restrict__ bv,
                     unsigned short* __restrict__ Qb, unsigned short* __restrict__ Kb,
                     unsigned short* __restrict__ Vb)
{
    __shared__ unsigned short Xs[128 * 40];  // 128x32 bf16, pad to 40
    __shared__ unsigned short Ws[64 * 40];   // 64x32 bf16 (row = out-feature n)

    const int z = blockIdx.z;
    const float* X    = (z == 0) ? query : (z == 1) ? key : value;
    const float* W    = (z == 0) ? Wq    : (z == 1) ? Wk  : Wv;
    const float* bias = (z == 0) ? bq    : (z == 1) ? bk  : bv;
    unsigned short* Out = (z == 0) ? Qb  : (z == 1) ? Kb  : Vb;

    const int m0 = blockIdx.x * 128;
    const int n0 = blockIdx.y * 64;
    const int tid  = threadIdx.x;
    const int w    = tid >> 5;
    const int lane = tid & 31;
    const int wr = w >> 1, wc = w & 1;
    const int ln = lane & 15, lh = lane >> 4;

    v8f acc[2][2];
#pragma unroll
    for (int i = 0; i < 2; ++i)
#pragma unroll
        for (int j = 0; j < 2; ++j) acc[i][j] = v8f{0.f,0.f,0.f,0.f,0.f,0.f,0.f,0.f};

    float4 xA[4], wA[2], xB[4], wB[2];

    auto load_regs = [&](int kk, float4* xr, float4* wr) {
#pragma unroll
        for (int it = 0; it < 4; ++it) {
            int task = tid + it * 256;
            int row = task >> 3, cg = (task & 7) * 4;
            xr[it] = *(const float4*)(X + (size_t)(m0 + row) * ND + kk + cg);
        }
#pragma unroll
        for (int it = 0; it < 2; ++it) {
            int task = tid + it * 256;
            int row = task >> 3, cg = (task & 7) * 4;
            wr[it] = *(const float4*)(W + (size_t)(n0 + row) * ND + kk + cg);
        }
    };
    auto store_regs = [&](const float4* xr, const float4* wr) {
#pragma unroll
        for (int it = 0; it < 4; ++it) {
            int task = tid + it * 256;
            int row = task >> 3, cg = (task & 7) * 4;
            unsigned short* d = &Xs[row * 40 + cg];
            d[0] = f2bf_u(xr[it].x); d[1] = f2bf_u(xr[it].y);
            d[2] = f2bf_u(xr[it].z); d[3] = f2bf_u(xr[it].w);
        }
#pragma unroll
        for (int it = 0; it < 2; ++it) {
            int task = tid + it * 256;
            int row = task >> 3, cg = (task & 7) * 4;
            unsigned short* d = &Ws[row * 40 + cg];
            d[0] = f2bf_u(wr[it].x); d[1] = f2bf_u(wr[it].y);
            d[2] = f2bf_u(wr[it].z); d[3] = f2bf_u(wr[it].w);
        }
    };
    auto compute = [&]() {
        v16bf a0 = load_a_frag(&Xs[(wr * 32 +  0) * 40], 40);
        v16bf a1 = load_a_frag(&Xs[(wr * 32 + 16) * 40], 40);
        v16bf b0 = load_b_frag(&Ws[(wc * 32 +  0) * 40], 40);
        v16bf b1 = load_b_frag(&Ws[(wc * 32 + 16) * 40], 40);
        acc[0][0] = wmma_bf16(a0, b0, acc[0][0]);
        acc[0][1] = wmma_bf16(a0, b1, acc[0][1]);
        acc[1][0] = wmma_bf16(a1, b0, acc[1][0]);
        acc[1][1] = wmma_bf16(a1, b1, acc[1][1]);
    };
    auto gstep = [&](int kk_next, const float4* xc, const float4* wc,
                     float4* xn, float4* wn, bool more) {
        __syncthreads();
        if (more) load_regs(kk_next, xn, wn);
        store_regs(xc, wc);
        __syncthreads();
        compute();
    };

    load_regs(0, xA, wA);
    for (int kk = 0; kk < ND; kk += 64) {
        gstep(kk + 32, xA, wA, xB, wB, true);
        gstep(kk + 64, xB, wB, xA, wA, kk + 64 < ND);
    }

    // store: head-major bf16 [b][h][s][dk]
#pragma unroll
    for (int tr = 0; tr < 2; ++tr)
#pragma unroll
        for (int tc = 0; tc < 2; ++tc) {
            int col = n0 + wc * 32 + tc * 16 + ln;
            float bvv = bias[col];
            int hh = col >> 6, dk = col & 63;
#pragma unroll
            for (int r = 0; r < 8; ++r) {
                int row = m0 + wr * 32 + tr * 16 + lh * 8 + r;
                int bb = row >> 11, ss = row & (NS - 1);
                float y = acc[tr][tc][r] + bvv;
                Out[((size_t)(bb * NH + hh) * NS + ss) * NDK + dk] = f2bf_u(y);
            }
        }
}

// =====================================================================
// Kernel 2: flash attention.  grid = (S/128, B*H), block = 256.
// K tiles: async double-buffer in LDS (prefetch tile i+1 during tile i).
// V tiles: register-prefetch one tile ahead, transposed into single buffer.
// =====================================================================
__global__ __launch_bounds__(256)
void attention_kernel(const unsigned short* __restrict__ Qb,
                      const unsigned short* __restrict__ Kb,
                      const unsigned short* __restrict__ Vb,
                      const int* __restrict__ mask,
                      unsigned short* __restrict__ Zb)
{
    __shared__ unsigned short Qs [128 * 72];      // queries (row pad 144B = 9*16)
    __shared__ unsigned short Ks [2 * 64 * 72];   // double-buffered key tiles
    __shared__ unsigned short Vts[64 * 72];       // transposed V [dv][kk]
    __shared__ unsigned short Ps [128 * 72];      // per-wave P strips

    const int q0 = blockIdx.x * 128;
    const int bh = blockIdx.y;
    const int b  = bh >> 4;
    const int h  = bh & 15;
    const int tid  = threadIdx.x;
    const int w    = tid >> 5;
    const int lane = tid & 31;
    const int ln = lane & 15, lh = lane >> 4;
    const size_t head_off = (size_t)bh * NS * NDK;
    const int NIT = NS / 64;                      // 32 key tiles

    // stage Q block once (async): 128 rows x 8 groups of 16B
#pragma unroll
    for (int it = 0; it < 4; ++it) {
        int task = tid + it * 256;
        int row = task >> 3, cg = (task & 7) * 8;
        async_ld_b128(lds_addr_of(&Qs[row * 72 + cg]),
                      Qb + head_off + (size_t)(q0 + row) * NDK + cg);
    }

    float m_i[8], l_i[8];
    v8f oacc[4];
#pragma unroll
    for (int r = 0; r < 8; ++r) { m_i[r] = -1e30f; l_i[r] = 0.f; }
#pragma unroll
    for (int t = 0; t < 4; ++t) oacc[t] = v8f{0.f,0.f,0.f,0.f,0.f,0.f,0.f,0.f};

    auto asyncK = [&](int i, int buf) {           // 2 async instr per wave
#pragma unroll
        for (int it = 0; it < 2; ++it) {
            int task = tid + it * 256;
            int row = task >> 3, cg = (task & 7) * 8;
            async_ld_b128(lds_addr_of(&Ks[buf * 64 * 72 + row * 72 + cg]),
                          Kb + head_off + (size_t)(i * 64 + row) * NDK + cg);
        }
    };
    auto loadV = [&](int i, uint2* vr) {
#pragma unroll
        for (int it = 0; it < 4; ++it) {
            int task = tid + it * 256;
            int row = task >> 4, cg = (task & 15) * 4;
            vr[it] = *(const uint2*)(Vb + head_off + (size_t)(i * 64 + row) * NDK + cg);
        }
    };
    auto storeVt = [&](const uint2* vr) {
#pragma unroll
        for (int it = 0; it < 4; ++it) {
            int task = tid + it * 256;
            int row = task >> 4, cg = (task & 15) * 4;
            Vts[(cg + 0) * 72 + row] = (unsigned short)(vr[it].x & 0xffff);
            Vts[(cg + 1) * 72 + row] = (unsigned short)(vr[it].x >> 16);
            Vts[(cg + 2) * 72 + row] = (unsigned short)(vr[it].y & 0xffff);
            Vts[(cg + 3) * 72 + row] = (unsigned short)(vr[it].y >> 16);
        }
    };

    auto attn_step = [&](int i, int kbuf, const uint2* vcur, uint2* vnext) {
        const int kt = i * 64;
        const bool more = (i + 1 < NIT);
        __syncthreads();                 // all waves done reading Ks[kbuf^1], Vts
        if (more) {
            asyncK(i + 1, kbuf ^ 1);     // prefetch next K tile (stays in flight)
            loadV(i + 1, vnext);         // prefetch next V tile into registers
        }
        storeVt(vcur);                   // current V tile -> LDS (transposed)
        if (more) asm volatile("s_wait_asynccnt 2" ::: "memory");  // tile i landed
        else      asm volatile("s_wait_asynccnt 0" ::: "memory");
        __syncthreads();                 // + dscnt wait: Vts visible

        const unsigned short* Kc = &Ks[kbuf * 64 * 72];

        // S = Q * K^T (16x64 per wave)
        v8f sacc[4];
#pragma unroll
        for (int t = 0; t < 4; ++t) sacc[t] = v8f{0.f,0.f,0.f,0.f,0.f,0.f,0.f,0.f};
#pragma unroll
        for (int kd = 0; kd < 64; kd += 32) {
            v16bf aq = load_a_frag(&Qs[(w * 16) * 72 + kd], 72);
#pragma unroll
            for (int t = 0; t < 4; ++t) {
                v16bf bk_ = load_b_frag(&Kc[(t * 16) * 72 + kd], 72);
                sacc[t] = wmma_bf16(aq, bk_, sacc[t]);
            }
        }

        // scale + mask + row max
        float tmax[8];
#pragma unroll
        for (int r = 0; r < 8; ++r) tmax[r] = -1e30f;
#pragma unroll
        for (int t = 0; t < 4; ++t) {
            int col = kt + t * 16 + ln;
            int mv  = mask[b * NS + col];
#pragma unroll
            for (int r = 0; r < 8; ++r) {
                float s = (mv != 0) ? sacc[t][r] * 0.125f : -1.0e9f;
                sacc[t][r] = s;
                tmax[r] = fmaxf(tmax[r], s);
            }
        }
#pragma unroll
        for (int r = 0; r < 8; ++r) {
            float v = tmax[r];
            v = fmaxf(v, __shfl_xor(v, 1, 32));
            v = fmaxf(v, __shfl_xor(v, 2, 32));
            v = fmaxf(v, __shfl_xor(v, 4, 32));
            v = fmaxf(v, __shfl_xor(v, 8, 32));
            tmax[r] = v;
        }
        float c[8], rs[8];
#pragma unroll
        for (int r = 0; r < 8; ++r) {
            float mn = fmaxf(m_i[r], tmax[r]);
            c[r] = __expf(m_i[r] - mn);
            m_i[r] = mn;
            rs[r] = 0.f;
        }
        // P = exp(S - m), bf16 into per-wave LDS strip
#pragma unroll
        for (int t = 0; t < 4; ++t) {
#pragma unroll
            for (int r = 0; r < 8; ++r) {
                float p = __expf(sacc[t][r] - m_i[r]);
                rs[r] += p;
                Ps[(w * 16 + lh * 8 + r) * 72 + t * 16 + ln] = f2bf_u(p);
            }
        }
#pragma unroll
        for (int r = 0; r < 8; ++r) {
            float v = rs[r];
            v += __shfl_xor(v, 1, 32);
            v += __shfl_xor(v, 2, 32);
            v += __shfl_xor(v, 4, 32);
            v += __shfl_xor(v, 8, 32);
            l_i[r] = l_i[r] * c[r] + v;
        }
#pragma unroll
        for (int t = 0; t < 4; ++t)
#pragma unroll
            for (int r = 0; r < 8; ++r) oacc[t][r] *= c[r];

        // wave-local LDS write->read ordering for Ps
        asm volatile("s_wait_dscnt 0" ::: "memory");

        // O += P * V
#pragma unroll
        for (int kd = 0; kd < 64; kd += 32) {
            v16bf ap = load_a_frag(&Ps[(w * 16) * 72 + kd], 72);
#pragma unroll
            for (int t = 0; t < 4; ++t) {
                v16bf bv_ = load_b_frag(&Vts[(t * 16) * 72 + kd], 72);
                oacc[t] = wmma_bf16(ap, bv_, oacc[t]);
            }
        }
    };

    uint2 vrA[4], vrB[4];
    asyncK(0, 0);
    loadV(0, vrA);
    for (int i = 0; i < NIT; i += 2) {
        attn_step(i,     0, vrA, vrB);
        attn_step(i + 1, 1, vrB, vrA);
    }

    // normalize and store Z bf16 [b][s][d]
#pragma unroll
    for (int r = 0; r < 8; ++r) {
        float inv = (l_i[r] > 0.f) ? 1.0f / l_i[r] : 0.f;
        int s = q0 + w * 16 + lh * 8 + r;
#pragma unroll
        for (int t = 0; t < 4; ++t) {
            int d = h * NDK + t * 16 + ln;
            Zb[((size_t)b * NS + s) * ND + d] = f2bf_u(oacc[t][r] * inv);
        }
    }
}

// =====================================================================
// Kernel 3: output projection.  out = Z * Wo^T + bo (fp32 out)
// Z: async double-buffered LDS tiles; Wo: register-prefetch + convert.
// grid = (M/128, N/64), block = 256
// =====================================================================
__global__ __launch_bounds__(256)
void out_proj_kernel(const unsigned short* __restrict__ Zb,
                     const float* __restrict__ Wo, const float* __restrict__ bo,
                     float* __restrict__ out)
{
    __shared__ unsigned short Xs2[2 * 128 * 40];  // double-buffered Z tiles
    __shared__ unsigned short Ws [64 * 40];

    const int m0 = blockIdx.x * 128;
    const int n0 = blockIdx.y * 64;
    const int tid  = threadIdx.x;
    const int w    = tid >> 5;
    const int lane = tid & 31;
    const int wr = w >> 1, wc = w & 1;
    const int ln = lane & 15, lh = lane >> 4;

    v8f acc[2][2];
#pragma unroll
    for (int i = 0; i < 2; ++i)
#pragma unroll
        for (int j = 0; j < 2; ++j) acc[i][j] = v8f{0.f,0.f,0.f,0.f,0.f,0.f,0.f,0.f};

    auto asyncZ = [&](int kk, int buf) {          // 2 async instr per wave
#pragma unroll
        for (int it = 0; it < 2; ++it) {
            int task = tid + it * 256;
            int row = task >> 2, cg = (task & 3) * 8;
            async_ld_b128(lds_addr_of(&Xs2[buf * 128 * 40 + row * 40 + cg]),
                          Zb + (size_t)(m0 + row) * ND + kk + cg);
        }
    };
    auto loadW = [&](int kk, float4* wr) {
#pragma unroll
        for (int it = 0; it < 2; ++it) {
            int task = tid + it * 256;
            int row = task >> 3, cg = (task & 7) * 4;
            wr[it] = *(const float4*)(Wo + (size_t)(n0 + row) * ND + kk + cg);
        }
    };
    auto storeW = [&](const float4* wr) {
#pragma unroll
        for (int it = 0; it < 2; ++it) {
            int task = tid + it * 256;
            int row = task >> 3, cg = (task & 7) * 4;
            unsigned short* d = &Ws[row * 40 + cg];
            d[0] = f2bf_u(wr[it].x); d[1] = f2bf_u(wr[it].y);
            d[2] = f2bf_u(wr[it].z); d[3] = f2bf_u(wr[it].w);
        }
    };
    auto compute = [&](int buf) {
        const unsigned short* Xc = &Xs2[buf * 128 * 40];
        v16bf a0 = load_a_frag(&Xc[(wr * 32 +  0) * 40], 40);
        v16bf a1 = load_a_frag(&Xc[(wr * 32 + 16) * 40], 40);
        v16bf b0 = load_b_frag(&Ws[(wc * 32 +  0) * 40], 40);
        v16bf b1 = load_b_frag(&Ws[(wc * 32 + 16) * 40], 40);
        acc[0][0] = wmma_bf16(a0, b0, acc[0][0]);
        acc[0][1] = wmma_bf16(a0, b1, acc[0][1]);
        acc[1][0] = wmma_bf16(a1, b0, acc[1][0]);
        acc[1][1] = wmma_bf16(a1, b1, acc[1][1]);
    };
    auto pstep = [&](int s, const float4* wcur, float4* wnext) {
        int buf = s & 1;
        bool more = (s + 1 < 32);
        __syncthreads();                  // buffer buf^1 free, Ws reads retired
        if (more) { asyncZ((s + 1) * 32, buf ^ 1); loadW((s + 1) * 32, wnext); }
        storeW(wcur);
        if (more) asm volatile("s_wait_asynccnt 2" ::: "memory");
        else      asm volatile("s_wait_asynccnt 0" ::: "memory");
        __syncthreads();
        compute(buf);
    };

    float4 wA[2], wB[2];
    asyncZ(0, 0);
    loadW(0, wA);
    for (int s = 0; s < 32; s += 2) {
        pstep(s,     wA, wB);
        pstep(s + 1, wB, wA);
    }

#pragma unroll
    for (int tr = 0; tr < 2; ++tr)
#pragma unroll
        for (int tc = 0; tc < 2; ++tc) {
            int col = n0 + wc * 32 + tc * 16 + ln;
            float bvv = bo[col];
#pragma unroll
            for (int r = 0; r < 8; ++r) {
                int row = m0 + wr * 32 + tr * 16 + lh * 8 + r;
                out[(size_t)row * ND + col] = acc[tr][tc][r] + bvv;
            }
        }
}

extern "C" void kernel_launch(void* const* d_in, const int* in_sizes, int n_in,
                              void* d_out, int out_size, void* d_ws, size_t ws_size,
                              hipStream_t stream) {
    const float* query = (const float*)d_in[0];
    const float* key   = (const float*)d_in[1];
    const float* value = (const float*)d_in[2];
    const int*   mask  = (const int*)d_in[3];
    const float* Wq = (const float*)d_in[4];
    const float* bq = (const float*)d_in[5];
    const float* Wk = (const float*)d_in[6];
    const float* bk = (const float*)d_in[7];
    const float* Wv = (const float*)d_in[8];
    const float* bv = (const float*)d_in[9];
    const float* Wo = (const float*)d_in[10];
    const float* bo = (const float*)d_in[11];
    float* out = (float*)d_out;

    // workspace: 4 bf16 tensors of B*S*D elements (8 MB each, 32 MB total)
    unsigned short* ws = (unsigned short*)d_ws;
    const size_t TEN = (size_t)NB * NS * ND;   // 4M elements
    unsigned short* Qb = ws;
    unsigned short* Kb = ws + TEN;
    unsigned short* Vb = ws + 2 * TEN;
    unsigned short* Zb = ws + 3 * TEN;

    qkv_proj_kernel<<<dim3((NB * NS) / 128, ND / 64, 3), 256, 0, stream>>>(
        query, key, value, Wq, bq, Wk, bk, Wv, bv, Qb, Kb, Vb);

    attention_kernel<<<dim3(NS / 128, NB * NH), 256, 0, stream>>>(
        Qb, Kb, Vb, mask, Zb);

    out_proj_kernel<<<dim3((NB * NS) / 128, ND / 64), 256, 0, stream>>>(
        Zb, Wo, bo, out);
}
